// WaveKANet_8976481648811
// MI455X (gfx1250) — matrix operand order repaired and checked by reading, not compiled, not run on hardware
//
#include <hip/hip_runtime.h>
#include <math.h>

typedef __attribute__((ext_vector_type(16))) __bf16 v16bf;
typedef __attribute__((ext_vector_type(8)))  float  v8f;

union Frag { uint4 u[2]; v16bf v; };

static __device__ __forceinline__ unsigned lds_addr_of(const void* p) {
  // low 32 bits of a flat shared pointer == LDS byte offset
  return (unsigned)(size_t)p;
}

// ---------------------------------------------------------------- zero
__global__ __launch_bounds__(256) void k_zero(unsigned* p, unsigned n) {
  unsigned i = blockIdx.x * 256u + threadIdx.x;
  unsigned stride = gridDim.x * 256u;
  for (; i < n; i += stride) p[i] = 0u;
}

// ---------------------------------------------------------------- weight pack
// wpk[o][tap(7x7)][c] bf16 ; fwb[o][c] bf16 ; biasbuf[384] f32
__global__ __launch_bounds__(256) void k_pack_w(const float* w3, const float* w5, const float* w7,
                                                const float* fusion_w,
                                                __bf16* wpk, __bf16* fwb, float* biasbuf,
                                                const float* b3, const float* b5, const float* b7) {
  const int blk = blockIdx.x, t = threadIdx.x;
  if (blk < 384) {
    const int o = blk, branch = o >> 7, ol = o & 127;
    const int p = branch + 1, k = 2 * p + 1, lo = 3 - p, hi = 3 + p;
    const float* src = (branch == 0) ? w3 : (branch == 1) ? w5 : w7;
    __bf16* dst = wpk + (size_t)o * 6272;
    for (int i = 0; i < 25; ++i) {
      int idx = i * 256 + t;
      if (idx < 6272) {
        int tap = idx >> 7, c = idx & 127, ky = tap / 7, kx = tap % 7;
        float v = 0.f;
        if (ky >= lo && ky <= hi && kx >= lo && kx <= hi)
          v = src[((size_t)(ol * 128 + c) * k + (ky - lo)) * k + (kx - lo)];
        dst[idx] = (__bf16)v;
      }
    }
  } else if (blk < 528) {
    int base = (blk - 384) * 1024;
    for (int i = 0; i < 4; ++i) {
      int idx = base + i * 256 + t;            // 144*1024 == 384*384
      fwb[idx] = (__bf16)fusion_w[idx];
    }
  } else {
    if (t < 384) {
      int br = t >> 7, ol = t & 127;
      biasbuf[t] = (br == 0) ? b3[ol] : (br == 1) ? b5[ol] : b7[ol];
    }
  }
}

// ---------------------------------------------------------------- x pack: NCHW f32 -> padded channel-last bf16
// xt[b][hp=70][wp=70][c=128] ; pad=3 border left zero (buffer pre-zeroed)
__global__ __launch_bounds__(256) void k_pack_x(const float* x, __bf16* xt) {
  __shared__ __bf16 T[64 * 132];               // [w][c], pitch 132 to dodge bank conflicts
  const int h = blockIdx.x, b = blockIdx.y, t = threadIdx.x;
  // load: all 128 channels of row h (coalesced along w), transpose into LDS
  for (int i = 0; i < 32; ++i) {
    int idx = i * 256 + t;                     // 8192 = 128c * 64w
    int c = idx >> 6, w = idx & 63;
    float v = x[(((size_t)b * 128 + c) * 64 + h) * 64 + w];
    T[w * 132 + c] = (__bf16)v;
  }
  __syncthreads();
  // store: channel-last, coalesced dwords (2 bf16 per dword)
  unsigned* out = (unsigned*)(xt + (((size_t)b * 70 + (h + 3)) * 70 + 3) * 128);
  for (int i = 0; i < 16; ++i) {
    int idx = i * 256 + t;                     // 4096 dwords = 64w * 64 dword
    int w = idx >> 6, c2 = idx & 63;
    unsigned short l, hh;
    __builtin_memcpy(&l,  &T[w * 132 + c2 * 2],     2);
    __builtin_memcpy(&hh, &T[w * 132 + c2 * 2 + 1], 2);
    out[(size_t)w * 64 + c2] = (unsigned)l | ((unsigned)hh << 16);
  }
}

// ---------------------------------------------------------------- conv implicit GEMM (WMMA bf16)
// grid (1024 m-tiles, 6 n-tiles), 256 thr = 8 waves; wave tile 32x32 (2x2 WMMA)
#define AS_PITCH 272   // 256B row + 16B pad -> conflict-free b128 frag reads
__global__ __launch_bounds__(256) void k_conv(const __bf16* __restrict__ xt,
                                              const __bf16* __restrict__ wpk,
                                              const float*  __restrict__ biasbuf,
                                              __bf16* __restrict__ fused,
                                              float* __restrict__ psum,
                                              float* __restrict__ pmax) {
  __shared__ __align__(16) unsigned char As[128 * AS_PITCH];   // 34 KB A tile
  __shared__ float Ps[2][2][16][4];
  __shared__ float Pm[2][2][16][4];

  const int t = threadIdx.x, lane = t & 31, wave = t >> 5;
  const int wm = wave & 3, wn = wave >> 2;
  const int mt = blockIdx.x, nt = blockIdx.y;
  const int m0 = mt << 7;
  const int b  = m0 >> 12;
  const int h0 = (m0 & 4095) >> 6;
  const int n0 = nt << 6;
  const int branch = n0 >> 7;
  const int p = branch + 1, tlo = 3 - p, thi = 3 + p;

  const char* xb = (const char*)(xt + (size_t)b * 70 * 70 * 128);
  const unsigned ldsA = lds_addr_of(&As[0]);
  const int lsel = (lane < 16) ? 0 : 16;
  const int l15  = lane & 15;

  v8f acc[2][2] = {};

  for (int ky = tlo; ky <= thi; ++ky) {
    for (int kx = tlo; kx <= thi; ++kx) {
      const int tap = ky * 7 + kx;
      const char* s0 = xb + ((size_t)(h0 + ky)     * 70 + kx) * 256; // 64w x 128c x 2B
      const char* s1 = xb + ((size_t)(h0 + 1 + ky) * 70 + kx) * 256;
      // async copy 32 KB tile -> LDS (row pitch AS_PITCH)
      #pragma unroll
      for (int i = 0; i < 8; ++i) {
        int lin = i * 256 + t;                 // 16B chunk id, 0..2047
        int off = lin * 16;
        const char* g = (off < 16384) ? (s0 + off) : (s1 + (off - 16384));
        unsigned lds = ldsA + (unsigned)((lin >> 4) * AS_PITCH + ((lin & 15) << 4));
        unsigned long long ga = (unsigned long long)(size_t)g;
        asm volatile("global_load_async_to_lds_b128 %0, %1, off"
                     :: "v"(lds), "v"(ga) : "memory");
      }
      asm volatile("s_wait_asynccnt 0" ::: "memory");
      __syncthreads();

      #pragma unroll
      for (int c0 = 0; c0 < 128; c0 += 32) {
        Frag afr[2];
        #pragma unroll
        for (int r = 0; r < 2; ++r) {
          int mrow = wm * 32 + r * 16 + l15;
          const uint4* pa = (const uint4*)&As[mrow * AS_PITCH + c0 * 2 + lsel];
          afr[r].u[0] = pa[0];
          afr[r].u[1] = pa[2];                 // +32B : K 16..23 / 24..31
        }
        Frag bfr[2];
        #pragma unroll
        for (int s = 0; s < 2; ++s) {
          int n = n0 + wn * 32 + s * 16 + l15;
          const uint4* pb = (const uint4*)(wpk + ((size_t)n * 49 + tap) * 128 + c0 + (lsel >> 1));
          bfr[s].u[0] = pb[0];
          bfr[s].u[1] = pb[2];                 // +32B : K 16..23 / 24..31
        }
        #pragma unroll
        for (int r = 0; r < 2; ++r)
          #pragma unroll
          for (int s = 0; s < 2; ++s)
            acc[r][s] = __builtin_amdgcn_wmma_f32_16x16x32_bf16(
                false, afr[r].v, false, bfr[s].v, (short)0, acc[r][s], false, false);
      }
      __syncthreads();
    }
  }

  // epilogue: bias, bf16 store, deterministic pool partials
  const int mlo = m0 + wm * 32 + ((lane >> 4) << 3);
  #pragma unroll
  for (int s = 0; s < 2; ++s) {
    const int col = n0 + wn * 32 + s * 16 + l15;
    const float bc = biasbuf[col];
    float lsum = 0.f, lmax = -3.0e38f;
    #pragma unroll
    for (int r = 0; r < 2; ++r) {
      #pragma unroll
      for (int v = 0; v < 8; ++v) {
        float val = acc[r][s][v] + bc;
        fused[(size_t)(mlo + r * 16 + v) * 384 + col] = (__bf16)val;
        lsum += val;
        lmax = fmaxf(lmax, val);
      }
    }
    lsum += __shfl_xor(lsum, 16, 32);
    lmax = fmaxf(lmax, __shfl_xor(lmax, 16, 32));
    if (lane < 16) { Ps[wn][s][lane][wm] = lsum; Pm[wn][s][lane][wm] = lmax; }
  }
  __syncthreads();
  if (wm == 0 && lane < 16) {
    #pragma unroll
    for (int s = 0; s < 2; ++s) {
      float ss = Ps[wn][s][lane][0] + Ps[wn][s][lane][1] + Ps[wn][s][lane][2] + Ps[wn][s][lane][3];
      float mm = fmaxf(fmaxf(Pm[wn][s][lane][0], Pm[wn][s][lane][1]),
                       fmaxf(Pm[wn][s][lane][2], Pm[wn][s][lane][3]));
      int col = n0 + wn * 32 + s * 16 + lane;
      psum[(size_t)mt * 384 + col] = ss;
      pmax[(size_t)mt * 384 + col] = mm;
    }
  }
}

// ---------------------------------------------------------------- channel attention
__global__ __launch_bounds__(256) void k_attn(const float* psum, const float* pmax,
                                              const float* w1, const float* w2, float* attn) {
  __shared__ float av[384], mv[384], sv[24];
  const int b = blockIdx.x, t = threadIdx.x;
  for (int e = t; e < 384; e += 256) {
    float s = 0.f, mx = -3.0e38f;
    for (int i = 0; i < 32; ++i) {
      s += psum[(size_t)(b * 32 + i) * 384 + e];
      mx = fmaxf(mx, pmax[(size_t)(b * 32 + i) * 384 + e]);
    }
    av[e] = s * (1.0f / 4096.0f);
    mv[e] = mx;
  }
  __syncthreads();
  if (t < 24) {
    float da = 0.f, dm = 0.f;
    for (int c = 0; c < 384; ++c) { float w = w1[t * 384 + c]; da += av[c] * w; dm += mv[c] * w; }
    sv[t] = fmaxf(da, 0.f) + fmaxf(dm, 0.f);
  }
  __syncthreads();
  for (int o = t; o < 384; o += 256) {
    float s = 0.f;
    for (int j = 0; j < 24; ++j) s += sv[j] * w2[o * 24 + j];
    attn[b * 384 + o] = 1.0f / (1.0f + expf(-s));
  }
}

// ---------------------------------------------------------------- apply gate in place (2 bf16 / dword)
__global__ __launch_bounds__(256) void k_scale(unsigned* fusedU, const float* attn) {
  size_t i = (size_t)blockIdx.x * 256 + threadIdx.x;       // total 131072*192 exactly
  unsigned m  = (unsigned)(i / 192);
  unsigned pr = (unsigned)(i % 192);
  unsigned b  = m >> 12;
  float a0 = attn[b * 384 + pr * 2];
  float a1 = attn[b * 384 + pr * 2 + 1];
  unsigned d = fusedU[i];
  float f0 = __uint_as_float((d & 0xffffu) << 16) * a0;
  float f1 = __uint_as_float(d & 0xffff0000u) * a1;
  __bf16 h0 = (__bf16)f0, h1 = (__bf16)f1;
  unsigned short s0, s1;
  __builtin_memcpy(&s0, &h0, 2); __builtin_memcpy(&s1, &h1, 2);
  fusedU[i] = (unsigned)s0 | ((unsigned)s1 << 16);
}

// ---------------------------------------------------------------- 1x1 fusion GEMM (WMMA) + LayerNorm
// block: 32 M x 384 N ; 8 waves: wm in {0,1} (16 rows), wn in {0..3} (96 cols = 6 subtiles)
#define OT_PITCH 388
__global__ __launch_bounds__(256) void k_fusion_ln(const __bf16* __restrict__ fused,
                                                   const __bf16* __restrict__ fwb,
                                                   const float* fb, const float* lg, const float* lb,
                                                   float* __restrict__ out) {
  __shared__ __align__(16) float Ot[32 * OT_PITCH];        // ~48.5 KB
  const int t = threadIdx.x, lane = t & 31, wave = t >> 5;
  const int wm = wave & 1, wn = wave >> 1;
  const int m0 = blockIdx.x << 5;
  const int lsel = (lane < 16) ? 0 : 16;
  const int l15  = lane & 15;

  v8f acc[6] = {};
  for (int c0 = 0; c0 < 384; c0 += 32) {
    Frag a;
    {
      int mrow = m0 + wm * 16 + l15;
      const uint4* pa = (const uint4*)((const char*)fused + ((size_t)mrow * 384 + c0) * 2 + lsel);
      a.u[0] = pa[0]; a.u[1] = pa[2];
    }
    #pragma unroll
    for (int s = 0; s < 6; ++s) {
      int n = wn * 96 + s * 16 + l15;
      const uint4* pb = (const uint4*)(fwb + (size_t)n * 384 + c0 + (lsel >> 1));
      Frag bf_; bf_.u[0] = pb[0]; bf_.u[1] = pb[2];
      acc[s] = __builtin_amdgcn_wmma_f32_16x16x32_bf16(
          false, a.v, false, bf_.v, (short)0, acc[s], false, false);
    }
  }
  const int rbase = wm * 16 + ((lane >> 4) << 3);
  #pragma unroll
  for (int s = 0; s < 6; ++s) {
    int col = wn * 96 + s * 16 + l15;
    float bias = fb[col];
    #pragma unroll
    for (int v = 0; v < 8; ++v)
      Ot[(rbase + v) * OT_PITCH + col] = acc[s][v] + bias;
  }
  __syncthreads();
  // LayerNorm over E=384, 4 rows per wave, coalesced f32 output
  for (int rr = 0; rr < 4; ++rr) {
    int row = wave * 4 + rr;
    float sum = 0.f, sq = 0.f;
    #pragma unroll
    for (int j = 0; j < 12; ++j) {
      float xv = Ot[row * OT_PITCH + j * 32 + lane];
      sum += xv; sq += xv * xv;
    }
    #pragma unroll
    for (int o = 16; o > 0; o >>= 1) {
      sum += __shfl_xor(sum, o, 32);
      sq  += __shfl_xor(sq,  o, 32);
    }
    float mu   = sum * (1.0f / 384.0f);
    float var  = sq  * (1.0f / 384.0f) - mu * mu;
    float rstd = rsqrtf(var + 1e-5f);
    size_t ob = (size_t)(m0 + row) * 384;
    #pragma unroll
    for (int j = 0; j < 12; ++j) {
      int e = j * 32 + lane;
      out[ob + e] = (Ot[row * OT_PITCH + e] - mu) * rstd * lg[e] + lb[e];
    }
  }
}

// ---------------------------------------------------------------- launch
extern "C" void kernel_launch(void* const* d_in, const int* in_sizes, int n_in,
                              void* d_out, int out_size, void* d_ws, size_t ws_size,
                              hipStream_t stream) {
  const float* x   = (const float*)d_in[0];
  const float* w3  = (const float*)d_in[1];
  const float* b3  = (const float*)d_in[2];
  const float* w5  = (const float*)d_in[3];
  const float* b5  = (const float*)d_in[4];
  const float* w7  = (const float*)d_in[5];
  const float* b7  = (const float*)d_in[6];
  const float* ca1 = (const float*)d_in[7];
  const float* ca2 = (const float*)d_in[8];
  const float* fw  = (const float*)d_in[9];
  const float* fb  = (const float*)d_in[10];
  const float* lg  = (const float*)d_in[11];
  const float* lbv = (const float*)d_in[12];

  char* ws = (char*)d_ws;
  const size_t SZ_XT  = (size_t)32 * 128 * 70 * 70 * 2;    // 40,140,800
  const size_t O_XT   = 0;
  const size_t O_WPK  = O_XT + SZ_XT;
  const size_t O_FWB  = O_WPK + (size_t)384 * 49 * 128 * 2;
  const size_t O_BIAS = O_FWB + (size_t)384 * 384 * 2;
  const size_t O_ATTN = O_BIAS + 1536;
  const size_t O_PSUM = O_ATTN + 49152;
  const size_t O_PMAX = O_PSUM + (size_t)1024 * 384 * 4;
  const size_t O_FUSE = O_PMAX + (size_t)1024 * 384 * 4;

  __bf16* xt    = (__bf16*)(ws + O_XT);
  __bf16* wpk   = (__bf16*)(ws + O_WPK);
  __bf16* fwb   = (__bf16*)(ws + O_FWB);
  float*  biasb = (float*)(ws + O_BIAS);
  float*  attn  = (float*)(ws + O_ATTN);
  float*  psum  = (float*)(ws + O_PSUM);
  float*  pmax  = (float*)(ws + O_PMAX);
  __bf16* fused = (__bf16*)(ws + O_FUSE);

  k_zero<<<4096, 256, 0, stream>>>((unsigned*)xt, (unsigned)(SZ_XT / 4));
  k_pack_w<<<529, 256, 0, stream>>>(w3, w5, w7, fw, wpk, fwb, biasb, b3, b5, b7);
  k_pack_x<<<dim3(64, 32), 256, 0, stream>>>(x, xt);
  k_conv<<<dim3(1024, 6), 256, 0, stream>>>(xt, wpk, biasb, fused, psum, pmax);
  k_attn<<<32, 256, 0, stream>>>(psum, pmax, ca1, ca2, attn);
  k_scale<<<98304, 256, 0, stream>>>((unsigned*)fused, attn);
  k_fusion_ln<<<4096, 256, 0, stream>>>(fused, fwb, fb, lg, lbv, (float*)d_out);
}
